// GraphSAGEModel_50371376447640
// MI455X (gfx1250) — matrix operand order repaired and checked by reading, not compiled, not run on hardware
//
#include <hip/hip_runtime.h>
#include <hip/hip_bf16.h>

typedef __attribute__((ext_vector_type(16))) _Float16 v16h;
typedef __attribute__((ext_vector_type(8)))  float    v8f;

#define N_NODES 100000
#define N_EDGES 1600000
#define N_LABEL 200000
#define DIM 64

// ---------------------------------------------------------------- zero fill
__global__ __launch_bounds__(256) void zero_kernel(float* __restrict__ p, int n) {
  int i = blockIdx.x * blockDim.x + threadIdx.x;
  int stride = gridDim.x * blockDim.x;
  for (; i < n; i += stride) p[i] = 0.0f;
}

// ------------------------------------------------ edge scatter (mean-agg num/den)
// 16 threads per edge, each thread owns 4 contiguous features (float4 gather,
// 4 global_atomic_add_f32 into the L2-resident accumulator).
__global__ __launch_bounds__(256) void scatter_kernel(const int* __restrict__ ei,
                                                      const float* __restrict__ xin,
                                                      float* __restrict__ agg,
                                                      float* __restrict__ cnt) {
  int t = blockIdx.x * blockDim.x + threadIdx.x;
  int e = t >> 4;
  if (e >= N_EDGES) return;
  int f = (t & 15) * 4;
  int src = ei[e];
  int dst = ei[N_EDGES + e];
  float4 v = *(const float4*)(xin + (size_t)src * DIM + f);
  float* a = agg + (size_t)dst * DIM + f;
  atomicAdd(a + 0, v.x);
  atomicAdd(a + 1, v.y);
  atomicAdd(a + 2, v.z);
  atomicAdd(a + 3, v.w);
  if (cnt != nullptr && (t & 15) == 0) atomicAdd(cnt + dst, 1.0f);
}

// ---------------------------------------------------------------- WMMA frags
// A fragment (16-bit A 16x32 layout): lane holds row (lane&15);
// halves 0..7  = K = kc + hi + 0..7,  halves 8..15 = K = kc + 16 + hi + 0..7
// where hi = 8 for lanes 16..31. Scale applied in f32 before f16 convert.
__device__ inline v16h frag_a(const float* __restrict__ rowp, float scale,
                              int kc, int lane) {
  int hi = (lane >> 4) * 8;
  float4 u0 = *(const float4*)(rowp + kc + hi);
  float4 u1 = *(const float4*)(rowp + kc + hi + 4);
  float4 u2 = *(const float4*)(rowp + kc + 16 + hi);
  float4 u3 = *(const float4*)(rowp + kc + 16 + hi + 4);
  v16h a;
  a[0]  = (_Float16)(u0.x * scale); a[1]  = (_Float16)(u0.y * scale);
  a[2]  = (_Float16)(u0.z * scale); a[3]  = (_Float16)(u0.w * scale);
  a[4]  = (_Float16)(u1.x * scale); a[5]  = (_Float16)(u1.y * scale);
  a[6]  = (_Float16)(u1.z * scale); a[7]  = (_Float16)(u1.w * scale);
  a[8]  = (_Float16)(u2.x * scale); a[9]  = (_Float16)(u2.y * scale);
  a[10] = (_Float16)(u2.z * scale); a[11] = (_Float16)(u2.w * scale);
  a[12] = (_Float16)(u3.x * scale); a[13] = (_Float16)(u3.y * scale);
  a[14] = (_Float16)(u3.z * scale); a[15] = (_Float16)(u3.w * scale);
  return a;
}

// B fragment (16-bit B 32x16 layout): lane holds column n = lane&15;
// K = kc + 8*(lane>=16) + 0..7 and kc + 16 + 8*(lane>=16) + 0..7. With B = W^T (W row-major [64][64]),
// B[k][n] = W[n][k] -> two runs of 8 contiguous floats of row n of W.
__device__ inline v16h frag_b(const float* __restrict__ W, int n, int kc, int lane) {
  const float* p = W + (size_t)n * DIM + kc + ((lane >> 4) * 8);
  float4 u0 = *(const float4*)(p);
  float4 u1 = *(const float4*)(p + 4);
  float4 u2 = *(const float4*)(p + 16);
  float4 u3 = *(const float4*)(p + 20);
  v16h b;
  b[0]  = (_Float16)u0.x; b[1]  = (_Float16)u0.y;
  b[2]  = (_Float16)u0.z; b[3]  = (_Float16)u0.w;
  b[4]  = (_Float16)u1.x; b[5]  = (_Float16)u1.y;
  b[6]  = (_Float16)u1.z; b[7]  = (_Float16)u1.w;
  b[8]  = (_Float16)u2.x; b[9]  = (_Float16)u2.y;
  b[10] = (_Float16)u2.z; b[11] = (_Float16)u2.w;
  b[12] = (_Float16)u3.x; b[13] = (_Float16)u3.y;
  b[14] = (_Float16)u3.z; b[15] = (_Float16)u3.w;
  return b;
}

// ------------------------------------------------ fused SAGE update per layer:
// out = relu( (agg/max(cnt,1)) @ Wl^T + bl + xin @ Wr^T )
// Block = 128 threads = 4 waves; block handles 16 nodes, wave w handles
// output columns [16w, 16w+16). 4 v_wmma_f32_16x16x32_f16 per wave.
__global__ __launch_bounds__(128) void sage_kernel(const float* __restrict__ agg,
                                                   const float* __restrict__ cnt,
                                                   const float* __restrict__ xin,
                                                   const float* __restrict__ Wl,
                                                   const float* __restrict__ bl,
                                                   const float* __restrict__ Wr,
                                                   float* __restrict__ out) {
  int lane = threadIdx.x & 31;
  int ct   = threadIdx.x >> 5;           // column tile 0..3
  int nb   = blockIdx.x;                 // 16-node block
  int row  = nb * 16 + (lane & 15);      // A row this lane feeds
  int n    = ct * 16 + (lane & 15);      // output column this lane feeds (B/D)

  float inv = 1.0f / fmaxf(cnt[row], 1.0f);
  const float* aggrow = agg + (size_t)row * DIM;
  const float* xrow   = xin + (size_t)row * DIM;

  v8f c = {};
#pragma unroll
  for (int kc = 0; kc < DIM; kc += 32) {
    v16h am = frag_a(aggrow, inv, kc, lane);
    v16h bm = frag_b(Wl, n, kc, lane);
    c = __builtin_amdgcn_wmma_f32_16x16x32_f16(false, am, false, bm,
                                               (short)0, c, false, false);
    v16h ax = frag_a(xrow, 1.0f, kc, lane);
    v16h bx = frag_b(Wr, n, kc, lane);
    c = __builtin_amdgcn_wmma_f32_16x16x32_f16(false, ax, false, bx,
                                               (short)0, c, false, false);
  }

  // C/D layout: VGPR r, lanes 0-15 -> M=r, lanes 16-31 -> M=r+8; N = lane&15.
  float bias = bl[n];
  int mbase = nb * 16 + (lane >> 4) * 8;
#pragma unroll
  for (int r = 0; r < 8; r++) {
    out[(size_t)(mbase + r) * DIM + n] = fmaxf(c[r] + bias, 0.0f);
  }
}

// ------------------------------------------------ edge scoring (wave per label)
// z = [h[s], h[d]] (128 f32); lane<16 covers h[s] float4s, lane>=16 covers h[d];
// dot with Ws, wave32 shfl_xor reduction, lane 0 adds bias and stores.
__global__ __launch_bounds__(256) void score_kernel(const int* __restrict__ eli,
                                                    const float* __restrict__ h,
                                                    const float* __restrict__ Ws,
                                                    const float* __restrict__ bs,
                                                    float* __restrict__ out) {
  int wid  = (blockIdx.x * blockDim.x + threadIdx.x) >> 5;
  int lane = threadIdx.x & 31;
  if (wid >= N_LABEL) return;
  int s = eli[wid];
  int d = eli[N_LABEL + wid];
  const float* row = (lane < 16) ? (h + (size_t)s * DIM) : (h + (size_t)d * DIM);
  int off = (lane & 15) * 4;
  float4 v = *(const float4*)(row + off);
  float4 w = *(const float4*)(Ws + (lane >> 4) * DIM + off);
  float dot = v.x * w.x + v.y * w.y + v.z * w.z + v.w * w.w;
#pragma unroll
  for (int sft = 16; sft >= 1; sft >>= 1) dot += __shfl_xor(dot, sft, 32);
  if (lane == 0) out[wid] = dot + bs[0];
}

// ---------------------------------------------------------------- launcher
extern "C" void kernel_launch(void* const* d_in, const int* in_sizes, int n_in,
                              void* d_out, int out_size, void* d_ws, size_t ws_size,
                              hipStream_t stream) {
  const float* x   = (const float*)d_in[0];
  const int*   ei  = (const int*)d_in[1];
  const int*   eli = (const int*)d_in[2];
  const float* Wl1 = (const float*)d_in[3];
  const float* bl1 = (const float*)d_in[4];
  const float* Wr1 = (const float*)d_in[5];
  const float* Wl2 = (const float*)d_in[6];
  const float* bl2 = (const float*)d_in[7];
  const float* Wr2 = (const float*)d_in[8];
  const float* Ws  = (const float*)d_in[9];
  const float* bs  = (const float*)d_in[10];
  float* out = (float*)d_out;

  // workspace layout (floats): agg[N*64] | cnt[N] | h1[N*64] | h2[N*64]
  float* agg = (float*)d_ws;
  float* cnt = agg + (size_t)N_NODES * DIM;
  float* h1  = cnt + N_NODES;
  float* h2  = h1 + (size_t)N_NODES * DIM;

  const int scat_blocks = (N_EDGES * 16) / 256;   // 100000
  const int sage_blocks = N_NODES / 16;           // 6250
  const int score_blocks = (N_LABEL * 32) / 256;  // 25000

  // ---- layer 1
  zero_kernel<<<2048, 256, 0, stream>>>(agg, N_NODES * DIM + N_NODES); // agg+cnt contiguous
  scatter_kernel<<<scat_blocks, 256, 0, stream>>>(ei, x, agg, cnt);
  sage_kernel<<<sage_blocks, 128, 0, stream>>>(agg, cnt, x, Wl1, bl1, Wr1, h1);

  // ---- layer 2 (cnt reused; only agg re-zeroed)
  zero_kernel<<<2048, 256, 0, stream>>>(agg, N_NODES * DIM);
  scatter_kernel<<<scat_blocks, 256, 0, stream>>>(ei, h1, agg, nullptr);
  sage_kernel<<<sage_blocks, 128, 0, stream>>>(agg, cnt, h1, Wl2, bl2, Wr2, h2);

  // ---- link scoring
  score_kernel<<<score_blocks, 256, 0, stream>>>(eli, h2, Ws, bs, out);
}
